// GATModel_53403623358888
// MI455X (gfx1250) — compile-verified, matrix operand order checked by reading, not run on hardware
//
#include <hip/hip_runtime.h>
#include <hip/hip_bf16.h>

// ---------------------------------------------------------------------------
// GAT autoencoder + 8-expert MoE for MI455X (gfx1250, wave32, WMMA).
// All GEMMs routed through ONE unguarded WMMA kernel (shapes padded on host):
//   - 256 threads = 8 waves/block, each wave owns a 16x64 C strip
//   - B tile (32k x 64n) staged cooperatively through LDS as f16 (shared by
//     all 8 waves; handles transB with coalesced loads either way)
//   - A fragments loaded as 4x global b128 per k-step (WMMA layout runs)
//   - fused bias + activation epilogue (none / lrelu / sigmoid)
// Graph segment-softmax via fp32 atomics (computed once, reused for the tied
// decoder propagation); BN via column reductions; LN via wave32 shuffles.
// ---------------------------------------------------------------------------

typedef __attribute__((ext_vector_type(16))) _Float16 v16h;
typedef __attribute__((ext_vector_type(8)))  float    v8f;

#define NN   60000
#define INC  256
#define HIDC 128
#define OUTC 64
#define KC   8
#define EC   900000
#define E1C  128
#define E2C  256
#define E3C  512
#define E4C  768
#define ZTP  96      // zt padded K (65 -> 96)

// ---------------------------------------------------------------------------
// Fast WMMA GEMM. Requirements (guaranteed by host-side padding):
//   M % 16 == 0, N % 64 == 0, K % 32 == 0, lda/ldb/ldc % 4 == 0,
//   A/B 16-byte aligned.  act: 0=none, 1=lrelu(0.01), 2=sigmoid
// ---------------------------------------------------------------------------
__global__ void __launch_bounds__(256)
gemm_wmma_fast(const float* __restrict__ A, const float* __restrict__ B,
               const float* __restrict__ bias, float* __restrict__ C,
               int M, int N, int K, int lda, int ldb, int ldc,
               int transB, int act)
{
    __shared__ _Float16 bsh[64 * 32];           // [n][k], 4 KB

    const int tid   = threadIdx.x;
    const int wave  = tid >> 5;
    const int lane  = tid & 31;
    const int tileM = blockIdx.y * 8 + wave;
    const int n0    = blockIdx.x * 64;
    const int r     = lane & 15;
    const int hi    = lane >> 4;
    const bool active = (tileM * 16 < M);       // wave-uniform
    const float* Arow = A + (size_t)(tileM * 16 + r) * lda;

    v8f acc[4] = {};

    for (int k0 = 0; k0 < K; k0 += 32) {
        // ---- cooperative B tile stage: 2048 elems, 8 per thread ----------
        __syncthreads();
        if (!transB) {
#pragma unroll
            for (int j = 0; j < 8; ++j) {
                int l = tid + 256 * j;          // n fastest -> coalesced
                int n = l & 63, k = l >> 6;
                bsh[n * 32 + k] = (_Float16)B[(size_t)(k0 + k) * ldb + (n0 + n)];
            }
        } else {
#pragma unroll
            for (int j = 0; j < 8; ++j) {
                int l = tid + 256 * j;          // k fastest -> coalesced
                int k = l & 31, n = l >> 5;
                bsh[n * 32 + k] = (_Float16)B[(size_t)(n0 + n) * ldb + (k0 + k)];
            }
        }
        __syncthreads();

        if (active) {
            // ---- A fragment: 16 halves = 4x b128 loads -------------------
            const float4 a0 = *(const float4*)(Arow + k0 + hi * 8);
            const float4 a1 = *(const float4*)(Arow + k0 + hi * 8 + 4);
            const float4 a2 = *(const float4*)(Arow + k0 + 16 + hi * 8);
            const float4 a3 = *(const float4*)(Arow + k0 + 16 + hi * 8 + 4);
            v16h af;
            af[0] = (_Float16)a0.x; af[1] = (_Float16)a0.y;
            af[2] = (_Float16)a0.z; af[3] = (_Float16)a0.w;
            af[4] = (_Float16)a1.x; af[5] = (_Float16)a1.y;
            af[6] = (_Float16)a1.z; af[7] = (_Float16)a1.w;
            af[8]  = (_Float16)a2.x; af[9]  = (_Float16)a2.y;
            af[10] = (_Float16)a2.z; af[11] = (_Float16)a2.w;
            af[12] = (_Float16)a3.x; af[13] = (_Float16)a3.y;
            af[14] = (_Float16)a3.z; af[15] = (_Float16)a3.w;

            // ---- 4 B fragments from LDS (contiguous 16B runs) ------------
#pragma unroll
            for (int s = 0; s < 4; ++s) {
                const _Float16* colp = bsh + (s * 16 + r) * 32 + hi * 8;
                v16h bf;
#pragma unroll
                for (int i = 0; i < 8; ++i) bf[i] = colp[i];
#pragma unroll
                for (int i = 0; i < 8; ++i) bf[8 + i] = colp[16 + i];
                acc[s] = __builtin_amdgcn_wmma_f32_16x16x32_f16(
                             false, af, false, bf, (short)0, acc[s], false, false);
            }
        }
    }

    if (!active) return;
    // ---- epilogue: unguarded, lane-coalesced stores ----------------------
#pragma unroll
    for (int s = 0; s < 4; ++s) {
        const int ocol = n0 + s * 16 + r;
        const float bsum = bias ? bias[ocol] : 0.0f;
#pragma unroll
        for (int j = 0; j < 8; ++j) {
            const int orow = tileM * 16 + hi * 8 + j;
            float val = acc[s][j] + bsum;
            if (act == 1)      val = (val >= 0.0f) ? val : 0.01f * val;
            else if (act == 2) val = 1.0f / (1.0f + expf(-val));
            C[(size_t)orow * ldc + ocol] = val;
        }
    }
}

// ---------------------------------------------------------------------------
__global__ void fill_kernel(float* __restrict__ p, long long n, float v)
{
    long long i = (long long)blockIdx.x * blockDim.x + threadIdx.x;
    if (i < n) p[i] = v;
}

// zero-padded 2-D copy: out[Kp][Np] <- in[Kin][Nin]
__global__ void pad2d_kernel(const float* __restrict__ in, float* __restrict__ out,
                             int Kin, int Nin, int Kp, int Np)
{
    long long i = (long long)blockIdx.x * blockDim.x + threadIdx.x;
    long long total = (long long)Kp * Np;
    if (i >= total) return;
    int rr = (int)(i / Np), cc = (int)(i % Np);
    out[i] = (rr < Kin && cc < Nin) ? in[(size_t)rr * Nin + cc] : 0.0f;
}

__global__ void att_logits_kernel(const float* __restrict__ x1,
                                  const float* __restrict__ as,
                                  const float* __restrict__ ad,
                                  float* __restrict__ a_src,
                                  float* __restrict__ a_dst, int Nr, int H)
{
    int i = blockIdx.x * blockDim.x + threadIdx.x;
    if (i >= Nr) return;
    const float* xr = x1 + (size_t)i * H;
    float s1 = 0.f, s2 = 0.f;
    for (int j = 0; j < H; ++j) { float v = xr[j]; s1 += v * as[j]; s2 += v * ad[j]; }
    a_src[i] = s1; a_dst[i] = s2;
}

// per-edge: a = sigmoid(a_src[src]+a_dst[dst]); segment-max via u32 bit trick
__global__ void edge_attn_kernel(const int* __restrict__ src, const int* __restrict__ dst,
                                 const float* __restrict__ a_src, const float* __restrict__ a_dst,
                                 float* __restrict__ a_e, unsigned int* __restrict__ amax_bits,
                                 long long E_)
{
    long long e = (long long)blockIdx.x * blockDim.x + threadIdx.x;
    if (e >= E_) return;
    float a = 1.0f / (1.0f + expf(-(a_src[src[e]] + a_dst[dst[e]])));
    a_e[e] = a;
    atomicMax(&amax_bits[dst[e]], __float_as_uint(a));
}

__global__ void edge_exp_kernel(const int* __restrict__ dst, const float* __restrict__ a_e,
                                const float* __restrict__ amax, float* __restrict__ ex_e,
                                float* __restrict__ denom, long long E_)
{
    long long e = (long long)blockIdx.x * blockDim.x + threadIdx.x;
    if (e >= E_) return;
    float ex = expf(a_e[e] - amax[dst[e]]);
    ex_e[e] = ex;
    atomicAdd(&denom[dst[e]], ex);
}

// scatter-add: out[dst] += xp[src] * w   (thread handles 4 dims of one edge)
__global__ void edge_aggr_kernel(const int* __restrict__ src, const int* __restrict__ dst,
                                 const float* __restrict__ xp, const float* __restrict__ ex_e,
                                 const float* __restrict__ denom, float* __restrict__ out,
                                 long long E_, int H)
{
    long long t = (long long)blockIdx.x * blockDim.x + threadIdx.x;
    const int perEdge = H >> 2;
    long long e = t / perEdge;
    int ch = (int)(t % perEdge);
    if (e >= E_) return;
    int s = src[e], d = dst[e];
    float w = ex_e[e] / (denom[d] + 1e-16f);
    const float* xs = xp + (size_t)s * H + ch * 4;
    float* od = out + (size_t)d * H + ch * 4;
#pragma unroll
    for (int j = 0; j < 4; ++j) atomicAdd(&od[j], xs[j] * w);
}

// column mean/var (biased) over Nr rows; one block per column
__global__ void __launch_bounds__(256)
col_stats_kernel(const float* __restrict__ X, float* __restrict__ mean,
                 float* __restrict__ var, int Nr, int C)
{
    __shared__ float ssum[256], ssq[256];
    int c = blockIdx.x;
    float s = 0.f, q = 0.f;
    for (int i = threadIdx.x; i < Nr; i += blockDim.x) {
        float v = X[(size_t)i * C + c];
        s += v; q += v * v;
    }
    ssum[threadIdx.x] = s; ssq[threadIdx.x] = q;
    __syncthreads();
    for (int off = 128; off > 0; off >>= 1) {
        if ((int)threadIdx.x < off) {
            ssum[threadIdx.x] += ssum[threadIdx.x + off];
            ssq[threadIdx.x]  += ssq[threadIdx.x + off];
        }
        __syncthreads();
    }
    if (threadIdx.x == 0) {
        float m = ssum[0] / (float)Nr;
        mean[c] = m;
        var[c]  = ssq[0] / (float)Nr - m * m;
    }
}

// y = act((x-mean)*rsqrt(var+eps)*g + b)   act: 0=none 1=elu 2=sigmoid
__global__ void bn_act_kernel(const float* __restrict__ X, const float* __restrict__ mean,
                              const float* __restrict__ var, const float* __restrict__ g,
                              const float* __restrict__ b, float* __restrict__ Y,
                              long long total, int C, int act)
{
    long long i = (long long)blockIdx.x * blockDim.x + threadIdx.x;
    if (i >= total) return;
    int c = (int)(i % C);
    float v = (X[i] - mean[c]) * rsqrtf(var[c] + 1e-5f) * g[c] + b[c];
    if (act == 1)      v = (v > 0.f) ? v : (expf(v) - 1.0f);
    else if (act == 2) v = 1.0f / (1.0f + expf(-v));
    Y[i] = v;
}

// row LayerNorm + leaky-relu; one wave32 per row (C <= 512)
__global__ void __launch_bounds__(256)
ln_lrelu_kernel(const float* __restrict__ X, const float* __restrict__ g,
                const float* __restrict__ b, float* __restrict__ Y, int Nr, int C)
{
    int wave = threadIdx.x >> 5, lane = threadIdx.x & 31;
    int row = blockIdx.x * 8 + wave;
    if (row >= Nr) return;
    const float* xr = X + (size_t)row * C;
    float vals[16];
    int cnt = 0;
    float s = 0.f;
    for (int c = lane; c < C; c += 32) { float v = xr[c]; vals[cnt++] = v; s += v; }
    for (int off = 16; off > 0; off >>= 1) s += __shfl_down(s, off, 32);
    s = __shfl(s, 0, 32);
    float m = s / (float)C;
    float q = 0.f;
    for (int i = 0; i < cnt; ++i) { float d = vals[i] - m; q += d * d; }
    for (int off = 16; off > 0; off >>= 1) q += __shfl_down(q, off, 32);
    q = __shfl(q, 0, 32);
    float inv = rsqrtf(q / (float)C + 1e-5f);
    int i = 0;
    for (int c = lane; c < C; c += 32) {
        float v = (vals[i++] - m) * inv * g[c] + b[c];
        Y[(size_t)row * C + c] = (v >= 0.f) ? v : 0.01f * v;
    }
}

// softmax over first 8 cols of a row with row-stride ld
__global__ void softmax8_kernel(const float* __restrict__ logits,
                                float* __restrict__ out, int Nr, int ld)
{
    int i = blockIdx.x * blockDim.x + threadIdx.x;
    if (i >= Nr) return;
    const float* lr = logits + (size_t)i * ld;
    float mx = lr[0];
    for (int j = 1; j < KC; ++j) mx = fmaxf(mx, lr[j]);
    float e[KC], s = 0.f;
    for (int j = 0; j < KC; ++j) { e[j] = expf(lr[j] - mx); s += e[j]; }
    float inv = 1.0f / s;
    for (int j = 0; j < KC; ++j) out[(size_t)i * KC + j] = e[j] * inv;
}

// o_t[i] = src[i*ld]  (extract column 0)
__global__ void col0_kernel(const float* __restrict__ src, float* __restrict__ dstp,
                            int Nr, int ld)
{
    int i = blockIdx.x * blockDim.x + threadIdx.x;
    if (i < Nr) dstp[i] = src[(size_t)i * ld];
}

// zt[N][96] = [z | t | zeros]
__global__ void concat_zt_kernel(const float* __restrict__ z, const float* __restrict__ t,
                                 float* __restrict__ zt, long long total)
{
    long long i = (long long)blockIdx.x * blockDim.x + threadIdx.x;
    if (i >= total) return;
    long long rr = i / ZTP;
    int cc = (int)(i % ZTP);
    zt[i] = (cc < OUTC) ? z[rr * OUTC + cc] : (cc == OUTC ? t[rr] : 0.0f);
}

// cp += c[:,k] * e4
__global__ void cp_accum_kernel(const float* __restrict__ e4, const float* __restrict__ c_full,
                                float* __restrict__ cp, long long total, int k)
{
    long long i = (long long)blockIdx.x * blockDim.x + threadIdx.x;
    if (i >= total) return;
    long long rr = i / E4C;
    cp[i] += c_full[rr * KC + k] * e4[i];
}

__global__ void copy_kernel(const float* __restrict__ src, float* __restrict__ dstp, long long n)
{
    long long i = (long long)blockIdx.x * blockDim.x + threadIdx.x;
    if (i < n) dstp[i] = src[i];
}

// ---------------------------------------------------------------------------
static inline void launch_gemm(const float* A, const float* B, const float* bias,
                               float* C, int M, int N, int K, int lda, int ldb,
                               int ldc, int transB, int act, hipStream_t s)
{
    int tm = (M + 15) / 16;
    dim3 grid(N / 64, (tm + 7) / 8);
    gemm_wmma_fast<<<grid, dim3(256), 0, s>>>(A, B, bias, C, M, N, K, lda, ldb, ldc, transB, act);
}

static inline dim3 grid1d(long long n, int b) { return dim3((unsigned)((n + b - 1) / b)); }

extern "C" void kernel_launch(void* const* d_in, const int* in_sizes, int n_in,
                              void* d_out, int out_size, void* d_ws, size_t ws_size,
                              hipStream_t stream)
{
    (void)in_sizes; (void)n_in; (void)out_size; (void)ws_size;

    const float* x       = (const float*)d_in[0];
    const int*   srcI    = (const int*)  d_in[1];
    const int*   dstI    = (const int*)  d_in[2];
    const float* W1      = (const float*)d_in[3];
    const float* att_src = (const float*)d_in[4];
    const float* att_dst = (const float*)d_in[5];
    const float* W2      = (const float*)d_in[6];
    const float* g1 = (const float*)d_in[7],  *b1 = (const float*)d_in[8];
    const float* g2 = (const float*)d_in[9],  *b2 = (const float*)d_in[10];
    const float* g3 = (const float*)d_in[11], *b3 = (const float*)d_in[12];
    const float* g4 = (const float*)d_in[13], *b4 = (const float*)d_in[14];
    const float* Wt1 = (const float*)d_in[15], *bt1 = (const float*)d_in[16];
    const float* Wt2 = (const float*)d_in[17], *bt2 = (const float*)d_in[18];
    const float* Wc1 = (const float*)d_in[19], *bc1 = (const float*)d_in[20];
    const float* Wc2 = (const float*)d_in[21], *bc2 = (const float*)d_in[22];
    const float* We1 = (const float*)d_in[23], *be1 = (const float*)d_in[24];
    const float* eg1 = (const float*)d_in[25], *eb1 = (const float*)d_in[26];
    const float* We2 = (const float*)d_in[27], *be2 = (const float*)d_in[28];
    const float* eg2 = (const float*)d_in[29], *eb2 = (const float*)d_in[30];
    const float* We3 = (const float*)d_in[31], *be3 = (const float*)d_in[32];
    const float* eg3 = (const float*)d_in[33], *eb3 = (const float*)d_in[34];
    const float* We4 = (const float*)d_in[35], *be4 = (const float*)d_in[36];
    const float* tf  = (const float*)d_in[37];

    float* out = (float*)d_out;
    float* o_recon = out;                                   // N*IN
    float* o_z     = o_recon + (size_t)NN * INC;            // N*OUT
    float* o_t     = o_z     + (size_t)NN * OUTC;           // N
    float* o_c     = o_t     + (size_t)NN;                  // N*K
    float* o_cp    = o_c     + (size_t)NN * KC;             // N*E4
    float* o_tf    = o_cp    + (size_t)NN * E4C;            // K*G

    // ---- workspace: persistent (edge + padded weights) + phase arena -----
    float* ws = (float*)d_ws;
    size_t off = 0;
    auto palloc = [&](size_t n) { float* p = ws + off; off += (n + 63) & ~(size_t)63; return p; };
    float* a_srcb = palloc(NN);
    float* a_dstb = palloc(NN);
    float* amaxb  = palloc(NN);
    float* denomb = palloc(NN);
    float* a_e    = palloc(EC);
    float* ex_e   = palloc(EC);
    float* meanb  = palloc(1024);
    float* varb   = palloc(1024);
    float* we1p   = palloc((size_t)KC * ZTP * E1C);   // 8 x [96][128]
    float* wc1p   = palloc(64 * 64);
    float* wc2p   = palloc(64 * 64);
    float* wt1p   = palloc(64 * 64);
    float* wt2p   = palloc(64 * 64);
    float* bc1p   = palloc(64);
    float* bc2p   = palloc(64);
    float* bt1p   = palloc(64);
    float* bt2p   = palloc(64);
    const size_t arena = off;

    // GAT phase arena
    size_t oG = arena;
    auto ga = [&](size_t n) { float* p = ws + oG; oG += (n + 63) & ~(size_t)63; return p; };
    float* x1   = ga((size_t)NN * HIDC);
    float* h    = ga((size_t)NN * HIDC);
    float* hb   = ga((size_t)NN * HIDC);
    float* ztmp = ga((size_t)NN * OUTC);
    float* z3   = ga((size_t)NN * HIDC);
    float* h2   = ga((size_t)NN * HIDC);
    float* h2b  = ga((size_t)NN * HIDC);
    float* rtmp = ga((size_t)NN * INC);

    // heads phase arena (overlays GAT arena)
    size_t oH = arena;
    auto ha = [&](size_t n) { float* p = ws + oH; oH += (n + 63) & ~(size_t)63; return p; };
    float* chb  = ha((size_t)NN * 64);
    float* thb  = ha((size_t)NN * 64);
    float* clgp = ha((size_t)NN * 64);
    float* ttmp = ha((size_t)NN * 64);

    // MoE phase arena (overlays heads arena)
    size_t oM = arena;
    auto ma = [&](size_t n) { float* p = ws + oM; oM += (n + 63) & ~(size_t)63; return p; };
    float* zt = ma((size_t)NN * ZTP);
    float* e1 = ma((size_t)NN * E1C);
    float* e2 = ma((size_t)NN * E2C);
    float* e3 = ma((size_t)NN * E3C);
    float* e4 = ma((size_t)NN * E4C);

    const int B = 256;

    // ---- host-side shape padding (tiny, once per call) -------------------
    for (int k = 0; k < KC; ++k)
        pad2d_kernel<<<grid1d((long long)ZTP * E1C, B), B, 0, stream>>>(
            We1 + (size_t)k * (OUTC + 1) * E1C, we1p + (size_t)k * ZTP * E1C,
            OUTC + 1, E1C, ZTP, E1C);
    pad2d_kernel<<<grid1d(64 * 64, B), B, 0, stream>>>(Wc1, wc1p, OUTC, 32, 64, 64);
    pad2d_kernel<<<grid1d(64 * 64, B), B, 0, stream>>>(Wc2, wc2p, 32, KC, 64, 64);
    pad2d_kernel<<<grid1d(64 * 64, B), B, 0, stream>>>(Wt1, wt1p, OUTC, 32, 64, 64);
    pad2d_kernel<<<grid1d(64 * 64, B), B, 0, stream>>>(Wt2, wt2p, 32, 1, 64, 64);
    pad2d_kernel<<<grid1d(64, B), B, 0, stream>>>(bc1, bc1p, 1, 32, 1, 64);
    pad2d_kernel<<<grid1d(64, B), B, 0, stream>>>(bc2, bc2p, 1, KC, 1, 64);
    pad2d_kernel<<<grid1d(64, B), B, 0, stream>>>(bt1, bt1p, 1, 32, 1, 64);
    pad2d_kernel<<<grid1d(64, B), B, 0, stream>>>(bt2, bt2p, 1, 1, 1, 64);

    // =============================== encode ===============================
    launch_gemm(x, W1, nullptr, x1, NN, HIDC, INC, INC, HIDC, HIDC, 0, 0, stream);
    att_logits_kernel<<<grid1d(NN, B), B, 0, stream>>>(x1, att_src, att_dst, a_srcb, a_dstb, NN, HIDC);
    fill_kernel<<<grid1d(NN, B), B, 0, stream>>>(amaxb, NN, 0.0f);
    fill_kernel<<<grid1d(NN, B), B, 0, stream>>>(denomb, NN, 0.0f);
    fill_kernel<<<grid1d((long long)NN * HIDC, B), B, 0, stream>>>(h, (long long)NN * HIDC, 0.0f);
    edge_attn_kernel<<<grid1d(EC, B), B, 0, stream>>>(srcI, dstI, a_srcb, a_dstb, a_e,
                                                      (unsigned int*)amaxb, EC);
    edge_exp_kernel<<<grid1d(EC, B), B, 0, stream>>>(dstI, a_e, amaxb, ex_e, denomb, EC);
    edge_aggr_kernel<<<grid1d((long long)EC * (HIDC / 4), B), B, 0, stream>>>(
        srcI, dstI, x1, ex_e, denomb, h, EC, HIDC);
    col_stats_kernel<<<dim3(HIDC), dim3(256), 0, stream>>>(h, meanb, varb, NN, HIDC);
    bn_act_kernel<<<grid1d((long long)NN * HIDC, B), B, 0, stream>>>(
        h, meanb, varb, g1, b1, hb, (long long)NN * HIDC, HIDC, 1);
    launch_gemm(hb, W2, nullptr, ztmp, NN, OUTC, HIDC, HIDC, OUTC, OUTC, 0, 0, stream);
    col_stats_kernel<<<dim3(OUTC), dim3(256), 0, stream>>>(ztmp, meanb, varb, NN, OUTC);
    bn_act_kernel<<<grid1d((long long)NN * OUTC, B), B, 0, stream>>>(
        ztmp, meanb, varb, g2, b2, o_z, (long long)NN * OUTC, OUTC, 0);

    // =============================== decode ===============================
    launch_gemm(o_z, W2, nullptr, z3, NN, HIDC, OUTC, OUTC, OUTC, HIDC, 1, 0, stream);
    fill_kernel<<<grid1d((long long)NN * HIDC, B), B, 0, stream>>>(h2, (long long)NN * HIDC, 0.0f);
    edge_aggr_kernel<<<grid1d((long long)EC * (HIDC / 4), B), B, 0, stream>>>(
        srcI, dstI, z3, ex_e, denomb, h2, EC, HIDC);
    col_stats_kernel<<<dim3(HIDC), dim3(256), 0, stream>>>(h2, meanb, varb, NN, HIDC);
    bn_act_kernel<<<grid1d((long long)NN * HIDC, B), B, 0, stream>>>(
        h2, meanb, varb, g3, b3, h2b, (long long)NN * HIDC, HIDC, 1);
    launch_gemm(h2b, W1, nullptr, rtmp, NN, INC, HIDC, HIDC, HIDC, INC, 1, 0, stream);
    col_stats_kernel<<<dim3(INC), dim3(256), 0, stream>>>(rtmp, meanb, varb, NN, INC);
    bn_act_kernel<<<grid1d((long long)NN * INC, B), B, 0, stream>>>(
        rtmp, meanb, varb, g4, b4, o_recon, (long long)NN * INC, INC, 2);

    // =============================== heads ================================
    launch_gemm(o_z, wc1p, bc1p, chb, NN, 64, OUTC, OUTC, 64, 64, 0, 1, stream);
    launch_gemm(chb, wc2p, bc2p, clgp, NN, 64, 64, 64, 64, 64, 0, 0, stream);
    softmax8_kernel<<<grid1d(NN, B), B, 0, stream>>>(clgp, o_c, NN, 64);
    launch_gemm(o_z, wt1p, bt1p, thb, NN, 64, OUTC, OUTC, 64, 64, 0, 1, stream);
    launch_gemm(thb, wt2p, bt2p, ttmp, NN, 64, 64, 64, 64, 64, 0, 2, stream);
    col0_kernel<<<grid1d(NN, B), B, 0, stream>>>(ttmp, o_t, NN, 64);

    // ========================== mixture of experts ========================
    concat_zt_kernel<<<grid1d((long long)NN * ZTP, B), B, 0, stream>>>(
        o_z, o_t, zt, (long long)NN * ZTP);
    fill_kernel<<<grid1d((long long)NN * E4C, B), B, 0, stream>>>(o_cp, (long long)NN * E4C, 0.0f);

    for (int k = 0; k < KC; ++k) {
        launch_gemm(zt, we1p + (size_t)k * ZTP * E1C, be1 + (size_t)k * E1C, e1,
                    NN, E1C, ZTP, ZTP, E1C, E1C, 0, 0, stream);
        ln_lrelu_kernel<<<grid1d((NN + 7) / 8, 1).x, 256, 0, stream>>>(
            e1, eg1 + (size_t)k * E1C, eb1 + (size_t)k * E1C, e1, NN, E1C);
        launch_gemm(e1, We2 + (size_t)k * E1C * E2C, be2 + (size_t)k * E2C, e2,
                    NN, E2C, E1C, E1C, E2C, E2C, 0, 0, stream);
        ln_lrelu_kernel<<<grid1d((NN + 7) / 8, 1).x, 256, 0, stream>>>(
            e2, eg2 + (size_t)k * E2C, eb2 + (size_t)k * E2C, e2, NN, E2C);
        launch_gemm(e2, We3 + (size_t)k * E2C * E3C, be3 + (size_t)k * E3C, e3,
                    NN, E3C, E2C, E2C, E3C, E3C, 0, 0, stream);
        ln_lrelu_kernel<<<grid1d((NN + 7) / 8, 1).x, 256, 0, stream>>>(
            e3, eg3 + (size_t)k * E3C, eb3 + (size_t)k * E3C, e3, NN, E3C);
        launch_gemm(e3, We4 + (size_t)k * E3C * E4C, be4 + (size_t)k * E4C, e4,
                    NN, E4C, E3C, E3C, E4C, E4C, 0, 2, stream);
        cp_accum_kernel<<<grid1d((long long)NN * E4C, B), B, 0, stream>>>(
            e4, o_c, o_cp, (long long)NN * E4C, k);
    }

    // type_features passthrough
    copy_kernel<<<grid1d((long long)KC * 256, B), B, 0, stream>>>(tf, o_tf, (long long)KC * 256);
}
